// UltraEnhancedGAT_Physics_39213051412489
// MI455X (gfx1250) — compile-verified, matrix operand order checked by reading, not run on hardware
//
#include <hip/hip_runtime.h>
#include <hip/hip_bf16.h>
#include <math.h>

// ---------------------------------------------------------------------------
// MI455X (gfx1250) implementation of the 2-layer edge-GAT + SE + pool + MLP.
// Dense GEMMs use V_WMMA_F32_16X16X4_F32 (f32 matrix cores: bandwidth-bound
// workload -> full precision is free).  Edge softmax/aggregation uses
// ordered-uint atomic max + f32 atomic adds (the true bandwidth bottleneck).
// ---------------------------------------------------------------------------

typedef __attribute__((ext_vector_type(2)))  float    v2f;
typedef __attribute__((ext_vector_type(8)))  float    v8f;
typedef __attribute__((ext_vector_type(16))) _Float16 v16h;

#define GEMM_MAX_K 512

// ---- monotonic float <-> uint encoding for atomic max on floats -----------
__device__ __forceinline__ unsigned fordu(float x) {
  unsigned u = __float_as_uint(x);
  return (u & 0x80000000u) ? ~u : (u | 0x80000000u);
}
__device__ __forceinline__ float funord(unsigned u) {
  return (u & 0x80000000u) ? __uint_as_float(u & 0x7fffffffu)
                           : __uint_as_float(~u);
}

// ---------------------------------------------------------------------------
// WMMA f32 GEMM:  C[M,N] = act(A[M,K] @ B[K,N] + bias)
// block = 128 threads = 4 waves; each wave owns one 16x16 output tile.
// A panel (16 x K) staged in LDS (coalesced); B read lane-coalesced.
// Inner loop is software-pipelined one step ahead so the ds/global loads for
// iteration k+1 overlap the v_wmma for iteration k; prefetch is issued
// unconditionally (speculative prefetch is silently dropped if OOB).
// act: 0 = none, 1 = relu, 2 = sigmoid
// ---------------------------------------------------------------------------
__global__ __launch_bounds__(128)
void gemm_wmma(const float* __restrict__ A, const float* __restrict__ Bm,
               const float* __restrict__ bias, float* __restrict__ C,
               int M, int K, int Ncol, int act)
{
  __shared__ float As[16 * GEMM_MAX_K];
  const int tid  = threadIdx.x;
  const int lane = tid & 31;
  const int wave = tid >> 5;
  const int m0   = blockIdx.x * 16;
  const int n0   = (blockIdx.y * 4 + wave) * 16;

  // cooperative, coalesced load of the A panel
  for (int idx = tid; idx < 16 * K; idx += 128) {
    int r = idx / K, c = idx - r * K;
    int gm = m0 + r; if (gm >= M) gm = M - 1;
    As[idx] = A[(size_t)gm * K + c];
  }
  __syncthreads();
  if (n0 >= Ncol) return;

  const int half = lane >> 4;   // selects K sub-pair per ISA A/B layout
  const int r    = lane & 15;   // A row within tile / B,C column within tile
  v8f acc = {};

#if __has_builtin(__builtin_amdgcn_wmma_f32_16x16x4_f32)
  const size_t bstep = (size_t)4 * Ncol;            // one k-step of B rows
  const float* Ap    = As + r * K + 2 * half;       // LDS, steps of 4
  const float* Brow  = Bm + (size_t)(2 * half) * Ncol + n0 + r;

  v2f a, b;
  a.x = Ap[0];  a.y = Ap[1];
  b.x = Brow[0]; b.y = Brow[Ncol];
  for (int k0 = 4; k0 < K; k0 += 4) {
    Ap   += 4;
    Brow += bstep;
    v2f an, bn;
    an.x = Ap[0];  an.y = Ap[1];          // next-iteration operands issue
    bn.x = Brow[0]; bn.y = Brow[Ncol];    // before the current WMMA waits
    __builtin_prefetch(Brow + 2 * bstep, 0, 1);  // speculative, unguarded
    acc = __builtin_amdgcn_wmma_f32_16x16x4_f32(false, a, false, b,
                                                (short)0, acc, false, false);
    a = an; b = bn;
  }
  acc = __builtin_amdgcn_wmma_f32_16x16x4_f32(false, a, false, b,
                                              (short)0, acc, false, false);
#else
  // fallback: codegen-confirmed f16 WMMA 16x16x32 (K is a multiple of 32 here)
  for (int k0 = 0; k0 < K; k0 += 32) {
    v16h a, b;
    for (int i = 0; i < 16; ++i) {
      int kk = k0 + (i < 8 ? i : i + 8) + (half ? 8 : 0);
      a[i] = (_Float16)As[r * K + kk];
      b[i] = (_Float16)Bm[(size_t)kk * Ncol + n0 + r];
    }
    acc = __builtin_amdgcn_wmma_f32_16x16x32_f16(false, a, false, b,
                                                 (short)0, acc, false, false);
  }
#endif

  // epilogue per 16x16 f32 C/D VGPR layout: col = n0 + (lane&15),
  // row = m0 + i + 8*(lane>=16)
  const int col = n0 + r;
  for (int i = 0; i < 8; ++i) {
    int row = m0 + i + 8 * half;
    if (row < M) {
      float y = acc[i];
      if (bias) y += bias[col];
      if (act == 1)      y = fmaxf(y, 0.f);
      else if (act == 2) y = 1.f / (1.f + __expf(-y));
      C[(size_t)row * Ncol + col] = y;
    }
  }
}

// ---------------------------------------------------------------------------
// Fold We @ att_edge into tiny per-edge-dim vectors (kills the [E,HC] tensor)
// aef1[d*4+h] = sum_c We1[d,h*128+c]*ae1[h,c];  aef2[d] = sum_c We2[d,c]*ae2[c]
// ---------------------------------------------------------------------------
__global__ void fuse_edge_w(const float* __restrict__ We1, const float* __restrict__ ae1,
                            const float* __restrict__ We2, const float* __restrict__ ae2,
                            float* __restrict__ aef1, float* __restrict__ aef2)
{
  int t = threadIdx.x;
  if (t < 64) {
    int d = t >> 2, h = t & 3;
    float s = 0.f;
    for (int c = 0; c < 128; ++c) s += We1[d * 512 + h * 128 + c] * ae1[h * 128 + c];
    aef1[d * 4 + h] = s;
  } else if (t < 80) {
    int d = t - 64;
    float s = 0.f;
    for (int c = 0; c < 128; ++c) s += We2[d * 128 + c] * ae2[c];
    aef2[d] = s;
  }
}

// per-edge: gate = sigmoid(logit[type]); ale{1,2} = (edge_attr*gate) @ aef{1,2}
__global__ void edge_prep(const float* __restrict__ edge_attr, const int* __restrict__ etype,
                          const float* __restrict__ logit,
                          const float* __restrict__ aef1, const float* __restrict__ aef2,
                          int E, float* __restrict__ ale1, float* __restrict__ ale2)
{
  int e = blockIdx.x * blockDim.x + threadIdx.x;
  if (e >= E) return;
  float gate = 1.f / (1.f + __expf(-logit[etype[e]]));
  float s0 = 0.f, s1 = 0.f, s2 = 0.f, s3 = 0.f, se = 0.f;
  for (int d = 0; d < 16; ++d) {
    float v = edge_attr[e * 16 + d] * gate;
    s0 += v * aef1[d * 4 + 0];
    s1 += v * aef1[d * 4 + 1];
    s2 += v * aef1[d * 4 + 2];
    s3 += v * aef1[d * 4 + 3];
    se += v * aef2[d];
  }
  ale1[e * 4 + 0] = s0; ale1[e * 4 + 1] = s1;
  ale1[e * 4 + 2] = s2; ale1[e * 4 + 3] = s3;
  ale2[e] = se;
}

// per-node attention projections: alsrc[n,h]=h[n,h,:]·a_src[h], same for dst.
// one block per node, one wave per head, wave32 shuffle reduction.
__global__ __launch_bounds__(128)
void node_att(const float* __restrict__ H, const float* __restrict__ asrc,
              const float* __restrict__ adst, int N, int heads, int hid,
              float* __restrict__ alsrc, float* __restrict__ aldst)
{
  int n = blockIdx.x;
  int wave = threadIdx.x >> 5, lane = threadIdx.x & 31;
  if (wave >= heads) return;
  const float* hp = H + (size_t)n * heads * hid + wave * hid;
  const float* as = asrc + wave * hid;
  const float* ad = adst + wave * hid;
  float ss = 0.f, sd = 0.f;
  for (int c = lane; c < hid; c += 32) {
    float v = hp[c];
    ss += v * as[c];
    sd += v * ad[c];
  }
  for (int m = 16; m; m >>= 1) {
    ss += __shfl_xor(ss, m);
    sd += __shfl_xor(sd, m);
  }
  if (lane == 0) {
    alsrc[n * heads + wave] = ss;
    aldst[n * heads + wave] = sd;
  }
}

// edge logits + leaky_relu + segment-max via ordered-uint atomic max
__global__ void edge_logit_max(const int* __restrict__ src, const int* __restrict__ dst,
                               const float* __restrict__ alsrc, const float* __restrict__ aldst,
                               float* __restrict__ al, unsigned* __restrict__ amax,
                               int E, int heads)
{
  int idx = blockIdx.x * blockDim.x + threadIdx.x;
  if (idx >= E * heads) return;
  int e = idx / heads, h = idx - e * heads;
  float v = alsrc[src[e] * heads + h] + aldst[dst[e] * heads + h] + al[idx];
  v = v > 0.f ? v : 0.2f * v;          // leaky_relu(0.2)
  al[idx] = v;
  atomicMax(&amax[dst[e] * heads + h], fordu(v));
}

// ex = exp(al - amax[dst]); den[dst] += ex (al overwritten in place)
__global__ void edge_exp_den(const int* __restrict__ dst, float* __restrict__ al,
                             const unsigned* __restrict__ amax, float* __restrict__ den,
                             int E, int heads)
{
  int idx = blockIdx.x * blockDim.x + threadIdx.x;
  if (idx >= E * heads) return;
  int e = idx / heads, h = idx - e * heads;
  float ex = __expf(al[idx] - funord(amax[dst[e] * heads + h]));
  al[idx] = ex;
  atomicAdd(&den[dst[e] * heads + h], ex);
}

// out[dst, h, c] += h1[src, h, c] * ex/(den[dst,h]+1e-16)  (main BW hotspot)
__global__ void aggregate(const int* __restrict__ src, const int* __restrict__ dst,
                          const float* __restrict__ H, const float* __restrict__ ex,
                          const float* __restrict__ den, float* __restrict__ out,
                          int E, int heads, int hid)
{
  int C = heads * hid;
  int idx = blockIdx.x * blockDim.x + threadIdx.x;
  if (idx >= E * C) return;
  int e = idx / C, j = idx - e * C, h = j / hid;
  float alpha = ex[e * heads + h] / (den[dst[e] * heads + h] + 1e-16f);
  atomicAdd(&out[(size_t)dst[e] * C + j], H[(size_t)src[e] * C + j] * alpha);
}

// BatchNorm stats: per-channel sum / sumsq (channels across lanes = coalesced)
__global__ __launch_bounds__(128)
void bn_stats(const float* __restrict__ X, const float* __restrict__ bias,
              int N, int C, float* __restrict__ sum, float* __restrict__ sq)
{
  int c = blockIdx.y * 128 + threadIdx.x;
  if (c >= C) return;
  int rows_per = (N + gridDim.x - 1) / gridDim.x;
  int r0 = blockIdx.x * rows_per;
  int r1 = r0 + rows_per; if (r1 > N) r1 = N;
  float b = bias[c], s = 0.f, q = 0.f;
  for (int r = r0; r < r1; ++r) {
    float v = X[(size_t)r * C + c] + b;
    s += v; q += v * v;
  }
  atomicAdd(&sum[c], s);
  atomicAdd(&sq[c], q);
}

// y = elu(gamma*(x+bias-m)*rsqrt(v+eps)+beta) [+ add]
__global__ void bn_apply_elu(const float* __restrict__ X, const float* __restrict__ bias,
                             const float* __restrict__ sum, const float* __restrict__ sq,
                             const float* __restrict__ g, const float* __restrict__ beta,
                             const float* __restrict__ add, float* __restrict__ out,
                             int N, int C)
{
  int idx = blockIdx.x * blockDim.x + threadIdx.x;
  if (idx >= N * C) return;
  int c = idx % C;
  float m  = sum[c] / (float)N;
  float v  = sq[c] / (float)N - m * m;
  float sc = g[c] * rsqrtf(v + 1e-5f);
  float y  = (X[idx] + bias[c] - m) * sc + beta[c];
  y = y > 0.f ? y : (__expf(y) - 1.f);
  if (add) y += add[idx];
  out[idx] = y;
}

__global__ void se_gate(float* __restrict__ x, const float* __restrict__ se, int n)
{
  int i = blockIdx.x * blockDim.x + threadIdx.x;
  if (i < n) x[i] *= se[i];
}

// graph pooling: mean-sum + ordered max + counts
__global__ void pool_kernel(const float* __restrict__ x2, const int* __restrict__ batch,
                            float* __restrict__ msum, unsigned* __restrict__ mmax,
                            float* __restrict__ cnt, int N, int C)
{
  int idx = blockIdx.x * blockDim.x + threadIdx.x;
  if (idx >= N * C) return;
  int n = idx / C, c = idx - n * C;
  int b = batch[n];
  float v = x2[idx];
  atomicAdd(&msum[b * C + c], v);
  atomicMax(&mmax[b * C + c], fordu(v));
  if (c == 0) atomicAdd(&cnt[b], 1.f);
}

// tiny head (B=8): g = relu(gf@ling), hcat=[mean,max,g,mean*max], 2-layer MLP
__global__ __launch_bounds__(256)
void final_mlp(const float* __restrict__ gf, const float* __restrict__ lw,
               const float* __restrict__ lb, const float* __restrict__ msum,
               const unsigned* __restrict__ mmax, const float* __restrict__ cnt,
               const float* __restrict__ w1, const float* __restrict__ b1,
               const float* __restrict__ w2, const float* __restrict__ b2,
               float* __restrict__ out)
{
  __shared__ float hcat[8 * 512];
  __shared__ float f1[8 * 128];
  int t = threadIdx.x;
  for (int i = t; i < 8 * 128; i += 256) {
    int b = i >> 7, c = i & 127;
    float mean = msum[i] / cnt[b];
    float mx   = funord(mmax[i]);
    float gg = lb[c];
    for (int k = 0; k < 32; ++k) gg += gf[b * 32 + k] * lw[k * 128 + c];
    gg = fmaxf(gg, 0.f);
    hcat[b * 512 + c]       = mean;
    hcat[b * 512 + 128 + c] = mx;
    hcat[b * 512 + 256 + c] = gg;
    hcat[b * 512 + 384 + c] = mean * mx;
  }
  __syncthreads();
  for (int i = t; i < 8 * 128; i += 256) {
    int b = i >> 7, c = i & 127;
    float s = b1[c];
    for (int j = 0; j < 512; ++j) s += hcat[b * 512 + j] * w1[j * 128 + c];
    f1[i] = fmaxf(s, 0.f);
  }
  __syncthreads();
  if (t < 8) {
    float s = b2[0];
    for (int c = 0; c < 128; ++c) s += f1[t * 128 + c] * w2[c];
    out[t] = s;
  }
}

// ---------------------------------------------------------------------------
extern "C" void kernel_launch(void* const* d_in, const int* in_sizes, int n_in,
                              void* d_out, int out_size, void* d_ws, size_t ws_size,
                              hipStream_t stream)
{
  (void)n_in; (void)out_size; (void)ws_size;
  const float* x          = (const float*)d_in[0];
  const float* edge_attr  = (const float*)d_in[1];
  const float* graph_feat = (const float*)d_in[2];
  const int*   edge_index = (const int*)d_in[3];
  const int*   etype      = (const int*)d_in[4];
  const int*   batch      = (const int*)d_in[5];
  const float* W1   = (const float*)d_in[6];
  const float* We1  = (const float*)d_in[7];
  const float* as1  = (const float*)d_in[8];
  const float* ad1  = (const float*)d_in[9];
  const float* ae1  = (const float*)d_in[10];
  const float* b1   = (const float*)d_in[11];
  const float* bn1g = (const float*)d_in[12];
  const float* bn1b = (const float*)d_in[13];
  const float* W2   = (const float*)d_in[14];
  const float* We2  = (const float*)d_in[15];
  const float* as2  = (const float*)d_in[16];
  const float* ad2  = (const float*)d_in[17];
  const float* ae2  = (const float*)d_in[18];
  const float* b2   = (const float*)d_in[19];
  const float* bn2g = (const float*)d_in[20];
  const float* bn2b = (const float*)d_in[21];
  const float* se_w1 = (const float*)d_in[22];
  const float* se_b1 = (const float*)d_in[23];
  const float* se_w2 = (const float*)d_in[24];
  const float* se_b2 = (const float*)d_in[25];
  const float* skip_w = (const float*)d_in[26];
  const float* skip_b = (const float*)d_in[27];
  const float* ling_w = (const float*)d_in[28];
  const float* ling_b = (const float*)d_in[29];
  const float* fin_w1 = (const float*)d_in[30];
  const float* fin_b1 = (const float*)d_in[31];
  const float* fin_w2 = (const float*)d_in[32];
  const float* fin_b2 = (const float*)d_in[33];
  const float* elogit = (const float*)d_in[34];

  const int N = in_sizes[0] / 128;   // 50000
  const int E = in_sizes[1] / 16;    // 400000
  const int* srcI = edge_index;
  const int* dstI = edge_index + E;

  // ---- workspace carve (floats) -------------------------------------------
  float* W = (float*)d_ws;
  size_t off = 0;
  auto take = [&](size_t n) { float* p = W + off; off += n; return p; };
  float* aef1   = take(64);
  float* aef2   = take(64);
  float* h1     = take((size_t)N * 512);   // layer-2 reuses this region
  float* ale1   = take((size_t)E * 4);
  float* ale2   = take((size_t)E);
  float* alsrc1 = take((size_t)N * 4);
  float* aldst1 = take((size_t)N * 4);
  unsigned* amax1 = (unsigned*)take((size_t)N * 4);
  float* den1   = take((size_t)N * 4);
  float* agg1   = take((size_t)N * 512);   // becomes x1; later reused for SE
  float* bnsum  = take(512);
  float* bnsq   = take(512);               // contiguous with bnsum
  float* alsrc2 = take(N);
  float* aldst2 = take(N);
  unsigned* amax2 = (unsigned*)take(N);
  float* den2   = take(N);
  float* msum   = take(1024);
  unsigned* mmax = (unsigned*)take(1024);  // contiguous with msum
  float* cnt    = take(16);                // contiguous with mmax

  // overlays (lifetimes verified: h1 dead after layer-1 aggregate,
  // x1(agg1) dead after skip GEMM)
  float* h2    = h1;
  float* agg2  = h1 + (size_t)N * 128;
  float* skipb = h1 + (size_t)N * 256;
  float* x2    = h1 + (size_t)N * 384;
  float* semid = agg1;
  float* seout = agg1 + (size_t)N * 64;

  const int MT = (N + 15) / 16;

  // ---- layer 1 ------------------------------------------------------------
  fuse_edge_w<<<1, 128, 0, stream>>>(We1, ae1, We2, ae2, aef1, aef2);
  edge_prep<<<(E + 255) / 256, 256, 0, stream>>>(edge_attr, etype, elogit,
                                                 aef1, aef2, E, ale1, ale2);
  gemm_wmma<<<dim3(MT, 8), 128, 0, stream>>>(x, W1, nullptr, h1, N, 128, 512, 0);
  node_att<<<N, 128, 0, stream>>>(h1, as1, ad1, N, 4, 128, alsrc1, aldst1);
  hipMemsetAsync(amax1, 0, (size_t)N * 4 * sizeof(float), stream);
  edge_logit_max<<<(E * 4 + 255) / 256, 256, 0, stream>>>(srcI, dstI, alsrc1, aldst1,
                                                          ale1, amax1, E, 4);
  hipMemsetAsync(den1, 0, (size_t)N * 4 * sizeof(float), stream);
  edge_exp_den<<<(E * 4 + 255) / 256, 256, 0, stream>>>(dstI, ale1, amax1, den1, E, 4);
  hipMemsetAsync(agg1, 0, (size_t)N * 512 * sizeof(float), stream);
  aggregate<<<(E * 512 + 255) / 256, 256, 0, stream>>>(srcI, dstI, h1, ale1, den1,
                                                       agg1, E, 4, 128);
  hipMemsetAsync(bnsum, 0, 1024 * sizeof(float), stream);
  bn_stats<<<dim3(128, 4), 128, 0, stream>>>(agg1, b1, N, 512, bnsum, bnsq);
  bn_apply_elu<<<(N * 512 + 255) / 256, 256, 0, stream>>>(agg1, b1, bnsum, bnsq,
                                                          bn1g, bn1b, nullptr, agg1, N, 512);

  // ---- layer 2 ------------------------------------------------------------
  gemm_wmma<<<dim3(MT, 2), 128, 0, stream>>>(agg1, W2, nullptr, h2, N, 512, 128, 0);
  node_att<<<N, 128, 0, stream>>>(h2, as2, ad2, N, 1, 128, alsrc2, aldst2);
  hipMemsetAsync(amax2, 0, (size_t)N * sizeof(float), stream);
  edge_logit_max<<<(E + 255) / 256, 256, 0, stream>>>(srcI, dstI, alsrc2, aldst2,
                                                      ale2, amax2, E, 1);
  hipMemsetAsync(den2, 0, (size_t)N * sizeof(float), stream);
  edge_exp_den<<<(E + 255) / 256, 256, 0, stream>>>(dstI, ale2, amax2, den2, E, 1);
  hipMemsetAsync(agg2, 0, (size_t)N * 128 * sizeof(float), stream);
  aggregate<<<(E * 128 + 255) / 256, 256, 0, stream>>>(srcI, dstI, h2, ale2, den2,
                                                       agg2, E, 1, 128);
  gemm_wmma<<<dim3(MT, 2), 128, 0, stream>>>(agg1, skip_w, skip_b, skipb, N, 512, 128, 0);
  hipMemsetAsync(bnsum, 0, 1024 * sizeof(float), stream);
  bn_stats<<<dim3(128, 1), 128, 0, stream>>>(agg2, b2, N, 128, bnsum, bnsq);
  bn_apply_elu<<<(N * 128 + 255) / 256, 256, 0, stream>>>(agg2, b2, bnsum, bnsq,
                                                          bn2g, bn2b, skipb, x2, N, 128);

  // ---- SE gating (activations fused into GEMM epilogues) ------------------
  gemm_wmma<<<dim3(MT, 1), 128, 0, stream>>>(x2, se_w1, se_b1, semid, N, 128, 64, 1);
  gemm_wmma<<<dim3(MT, 2), 128, 0, stream>>>(semid, se_w2, se_b2, seout, N, 64, 128, 2);
  se_gate<<<(N * 128 + 255) / 256, 256, 0, stream>>>(x2, seout, N * 128);

  // ---- pooling + head -----------------------------------------------------
  hipMemsetAsync(msum, 0, (1024 + 1024 + 16) * sizeof(float), stream);
  pool_kernel<<<(N * 128 + 255) / 256, 256, 0, stream>>>(x2, batch, msum, mmax, cnt, N, 128);
  final_mlp<<<1, 256, 0, stream>>>(graph_feat, ling_w, ling_b, msum, mmax, cnt,
                                   fin_w1, fin_b1, fin_w2, fin_b2, (float*)d_out);
}
